// RelativeScaledDotProductAttention_42073499632234
// MI455X (gfx1250) — matrix operand order repaired and checked by reading, not compiled, not run on hardware
//
#include <hip/hip_runtime.h>

// ---------------------------------------------------------------------------
// RelativeScaledDotProductAttention  (B=32, L=1024, D=128, fp32)
//   qk      = einsum('bqd,bkd->bqk', Q, K)
//   rel_qk  = einsum('bqd,qkd->bqk', Q, a_key)
//   score   = (qk + rel_qk)/sqrt(D);  score[mask] = -1e18
//   attn    = softmax(score, -1)
//   out     = einsum('bqk,bkd->bqd', attn, V) + einsum('bqk,qkd->bqd', attn, a_value)
// Outputs (concatenated): out [B,L,D] then attn [B,L,L]
//
// Bandwidth-bound on a_key/a_value (512MB each, streamed once, NT hint).
// All GEMMs on v_wmma_f32_16x16x32_bf16 (f32 accumulate); f32->bf16 pairs
// packed with v_perm_b32 (3 VALU per 2 elements).
// ---------------------------------------------------------------------------

#define B_  32
#define L_  1024
#define D_  128
#define INV_SCALE 0.088388347648318447f   // 1/sqrt(128)

typedef __attribute__((ext_vector_type(16))) __bf16 v16bf;
typedef __attribute__((ext_vector_type(2)))  __bf16 v2bf;
typedef __attribute__((ext_vector_type(8)))  float  v8f;
typedef __attribute__((ext_vector_type(4)))  float  v4f;

union ABu { unsigned int w[8]; v16bf v; };

// pack two f32 -> two bf16 in one dword (element 0 = a in low 16 bits)
__device__ __forceinline__ unsigned int pk2(float a, float b) {
#if __has_builtin(__builtin_amdgcn_cvt_pk_bf16_f32)
  v2bf p = __builtin_amdgcn_cvt_pk_bf16_f32(a, b);
  return __builtin_bit_cast(unsigned int, p);
#else
  // round-half-up then take high 16 bits of each; one v_perm_b32 packs both.
  unsigned ua = __float_as_uint(a) + 0x8000u;
  unsigned ub = __float_as_uint(b) + 0x8000u;
  // v_perm_b32 pool: bytes 0-3 = S1(ub), bytes 4-7 = S0(ua).
  // result bytes (LSB..MSB) = {ua[2], ua[3], ub[2], ub[3]} -> sel 0x03020706
  return __builtin_amdgcn_perm(ua, ub, 0x03020706u);
#endif
}

__device__ __forceinline__ v8f wmma_bf16(v16bf a, v16bf b, v8f c) {
  // 8 args: (neg_a, A, neg_b, B, c_mod, C, reuse_a, reuse_b)
  return __builtin_amdgcn_wmma_f32_16x16x32_bf16(false, a, false, b, (short)0, c, false, false);
}

// ---- A fragment (16x32 bf16, MxK). lane: m = lane&15, khalf = lane>>4.
// Element e -> K = khalf*16 + e: 16 consecutive K.
// `row` points at the 32-float K-chunk of this lane's M row.
__device__ __forceinline__ v16bf load_a_frag(const float* __restrict__ row, int khalf) {
  const float* p = row + khalf * 16;
  ABu t;
#pragma unroll
  for (int h = 0; h < 2; ++h) {              // h: K run (base h*8)
    v4f x0 = *(const v4f*)(p + h * 8);
    v4f x1 = *(const v4f*)(p + h * 8 + 4);
    t.w[4 * h + 0] = pk2(x0.x, x0.y);
    t.w[4 * h + 1] = pk2(x0.z, x0.w);
    t.w[4 * h + 2] = pk2(x1.x, x1.y);
    t.w[4 * h + 3] = pk2(x1.z, x1.w);
  }
  return t.v;
}

// ---- B fragment (32x16 bf16, KxN), lane: n = lane&15, K = khalf*16 + e.
// `p` points at 16 consecutive K-values for this lane's N column.
__device__ __forceinline__ v16bf load_b_frag_contig(const float* __restrict__ p) {
  ABu t;
#pragma unroll
  for (int i = 0; i < 4; ++i) {
    v4f x = *(const v4f*)(p + 4 * i);
    t.w[2 * i + 0] = pk2(x.x, x.y);
    t.w[2 * i + 1] = pk2(x.z, x.w);
  }
  return t.v;
}

// Same, non-temporal (stream-once tables)
__device__ __forceinline__ v16bf load_b_frag_contig_nt(const float* __restrict__ p) {
  ABu t;
#pragma unroll
  for (int i = 0; i < 4; ++i) {
    v4f x = __builtin_nontemporal_load((const v4f*)(p + 4 * i));
    t.w[2 * i + 0] = pk2(x.x, x.y);
    t.w[2 * i + 1] = pk2(x.z, x.w);
  }
  return t.v;
}

// ---- B fragment with row stride (column-slice of a row-major [K x N] block):
// element e reads p[e * stride]; lanes are coalesced across n = lane&15.
__device__ __forceinline__ v16bf load_b_frag_strided_nt(const float* __restrict__ p, int stride) {
  ABu t;
#pragma unroll
  for (int i = 0; i < 8; ++i) {
    float a = __builtin_nontemporal_load(p + (2 * i) * stride);
    float b = __builtin_nontemporal_load(p + (2 * i + 1) * stride);
    t.w[i] = pk2(a, b);
  }
  return t.v;
}
__device__ __forceinline__ v16bf load_b_frag_strided(const float* __restrict__ p, int stride) {
  ABu t;
#pragma unroll
  for (int i = 0; i < 8; ++i) {
    t.w[i] = pk2(p[(2 * i) * stride], p[(2 * i + 1) * stride]);
  }
  return t.v;
}

// ---------------------------------------------------------------------------
// Kernel R: rel[b,q,k] = sum_d Q[b,q,d]*a_key[q,k,d]   (raw, unscaled)
// grid = (L/64 n-strips, L q) , block = 128 (4 waves, one 16-col n-tile each)
// ---------------------------------------------------------------------------
__global__ void __launch_bounds__(128)
rel_scores_kernel(const float* __restrict__ q, const float* __restrict__ a_key,
                  float* __restrict__ rel_out) {
  const int qi    = blockIdx.y;
  const int wave  = threadIdx.x >> 5;
  const int lane  = threadIdx.x & 31;
  const int m     = lane & 15;
  const int khalf = lane >> 4;
  const int ncol0 = (blockIdx.x * 4 + wave) * 16;  // k-column base, 0..1023

  // A: Q[:, qi, :]  rows = batch (M=32 -> 2 m-tiles), K = D = 4 chunks of 32
  v16bf a[2][4];
#pragma unroll
  for (int mt = 0; mt < 2; ++mt) {
    const float* src = q + ((size_t)(mt * 16 + m) * L_ + qi) * D_;
#pragma unroll
    for (int kc = 0; kc < 4; ++kc) a[mt][kc] = load_a_frag(src + kc * 32, khalf);
  }

  v8f acc[2] = {};
#pragma unroll
  for (int kc = 0; kc < 4; ++kc) {
    // B[d][n] = a_key[qi, ncol0+n, d] : 16 consecutive d per lane, NT stream
    const float* bsrc = a_key + ((size_t)qi * L_ + (ncol0 + m)) * D_ + kc * 32 + khalf * 16;
    v16bf bf = load_b_frag_contig_nt(bsrc);
    acc[0] = wmma_bf16(a[0][kc], bf, acc[0]);
    acc[1] = wmma_bf16(a[1][kc], bf, acc[1]);
  }

#pragma unroll
  for (int mt = 0; mt < 2; ++mt)
#pragma unroll
    for (int r = 0; r < 8; ++r) {
      const int b = mt * 16 + r + khalf * 8;
      rel_out[((size_t)b * L_ + qi) * L_ + ncol0 + m] = acc[mt][r];
    }
}

// ---------------------------------------------------------------------------
// Kernel S: score = (Q@K^T + rel)/sqrt(D), mask, softmax -> attn (in place)
// grid = (L/16 q-tiles, B) , block = 256 (8 waves)
// ---------------------------------------------------------------------------
__global__ void __launch_bounds__(256)
qk_softmax_kernel(const float* __restrict__ q, const float* __restrict__ k,
                  const unsigned char* __restrict__ mask,
                  float* __restrict__ attn /* in: raw rel, out: attn */) {
  __shared__ float sc[16][L_];
  const int b     = blockIdx.y;
  const int qt    = blockIdx.x;
  const int wave  = threadIdx.x >> 5;
  const int lane  = threadIdx.x & 31;
  const int m     = lane & 15;
  const int khalf = lane >> 4;

  // A: Q[b, qt*16 + m, :]  (shared across this wave's n-tiles)
  v16bf a[4];
  {
    const float* src = q + ((size_t)b * L_ + qt * 16 + m) * D_;
#pragma unroll
    for (int kc = 0; kc < 4; ++kc) a[kc] = load_a_frag(src + kc * 32, khalf);
  }

  for (int nt = wave; nt < L_ / 16; nt += 8) {
    v8f acc = {};
    const float* ksrc = k + ((size_t)b * L_ + nt * 16 + m) * D_ + khalf * 16;
#pragma unroll
    for (int kc = 0; kc < 4; ++kc)
      acc = wmma_bf16(a[kc], load_b_frag_contig(ksrc + kc * 32), acc);

#pragma unroll
    for (int r = 0; r < 8; ++r) {
      const int row = r + khalf * 8;
      const int col = nt * 16 + m;
      const size_t gq = (size_t)b * L_ + qt * 16 + row;
      float s = (acc[r] + attn[gq * L_ + col]) * INV_SCALE;
      if (__builtin_nontemporal_load(mask + gq * L_ + col)) s = -1e18f;
      sc[row][col] = s;
    }
  }
  __syncthreads();

  // softmax: each wave owns 2 rows
#pragma unroll
  for (int rr = 0; rr < 2; ++rr) {
    const int row = wave * 2 + rr;
    float mx = -3.4e38f;
    for (int c = lane; c < L_; c += 32) mx = fmaxf(mx, sc[row][c]);
#pragma unroll
    for (int off = 16; off; off >>= 1) mx = fmaxf(mx, __shfl_xor(mx, off, 32));
    float sum = 0.f;
    for (int c = lane; c < L_; c += 32) { float e = __expf(sc[row][c] - mx); sc[row][c] = e; sum += e; }
#pragma unroll
    for (int off = 16; off; off >>= 1) sum += __shfl_xor(sum, off, 32);
    const float inv = 1.0f / sum;
    const size_t gq = (size_t)b * L_ + qt * 16 + row;
    for (int c = lane; c < L_; c += 32) attn[gq * L_ + c] = sc[row][c] * inv;
  }
}

// ---------------------------------------------------------------------------
// Kernel O1: out[b] = attn[b] @ V[b]
// grid = (L/16 q-tiles, B) , block = 256 (8 waves, one 16-wide d-tile each)
// ---------------------------------------------------------------------------
__global__ void __launch_bounds__(256)
av_kernel(const float* __restrict__ attn, const float* __restrict__ v,
          float* __restrict__ out) {
  const int b     = blockIdx.y;
  const int qt    = blockIdx.x;
  const int wave  = threadIdx.x >> 5;
  const int lane  = threadIdx.x & 31;
  const int m     = lane & 15;
  const int khalf = lane >> 4;

  v8f acc = {};
  const float* arow = attn + ((size_t)b * L_ + qt * 16 + m) * L_;
  for (int kc = 0; kc < L_ / 32; ++kc) {
    v16bf ta = load_a_frag(arow + kc * 32, khalf);
    // B[kk][n] = V[b, kc*32+kk, wave*16+n]
    const float* bsrc = v + ((size_t)b * L_ + kc * 32 + khalf * 16) * D_ + wave * 16 + m;
    acc = wmma_bf16(ta, load_b_frag_strided(bsrc, D_), acc);
  }
#pragma unroll
  for (int r = 0; r < 8; ++r)
    out[((size_t)b * L_ + qt * 16 + r + khalf * 8) * D_ + wave * 16 + m] = acc[r];
}

// ---------------------------------------------------------------------------
// Kernel O2: out[:,q,:] += attn[:,q,:] @ a_value[q]   (one block per q)
// grid = L , block = 256 (8 waves, one 16-wide d-tile each; 2 m-tiles for B=32)
// ---------------------------------------------------------------------------
__global__ void __launch_bounds__(256)
rel_av_kernel(const float* __restrict__ attn, const float* __restrict__ a_value,
              float* __restrict__ out) {
  const int qi    = blockIdx.x;
  const int wave  = threadIdx.x >> 5;
  const int lane  = threadIdx.x & 31;
  const int m     = lane & 15;
  const int khalf = lane >> 4;

  v8f acc[2] = {};
  for (int kc = 0; kc < L_ / 32; ++kc) {
    // B[kk][n] = a_value[qi, kc*32+kk, wave*16+n] : stream-once, NT
    const float* bsrc = a_value + ((size_t)qi * L_ + kc * 32 + khalf * 16) * D_ + wave * 16 + m;
    const v16bf bf = load_b_frag_strided_nt(bsrc, D_);
#pragma unroll
    for (int mt = 0; mt < 2; ++mt) {
      const float* arow = attn + ((size_t)(mt * 16 + m) * L_ + qi) * L_;
      acc[mt] = wmma_bf16(load_a_frag(arow + kc * 32, khalf), bf, acc[mt]);
    }
  }
#pragma unroll
  for (int mt = 0; mt < 2; ++mt)
#pragma unroll
    for (int r = 0; r < 8; ++r) {
      const size_t o = ((size_t)(mt * 16 + r + khalf * 8) * L_ + qi) * D_ + wave * 16 + m;
      out[o] += acc[mt][r];
    }
}

// ---------------------------------------------------------------------------
extern "C" void kernel_launch(void* const* d_in, const int* in_sizes, int n_in,
                              void* d_out, int out_size, void* d_ws, size_t ws_size,
                              hipStream_t stream) {
  const float*         q       = (const float*)d_in[0];
  const float*         k       = (const float*)d_in[1];
  const float*         v       = (const float*)d_in[2];
  const float*         a_key   = (const float*)d_in[3];
  const float*         a_value = (const float*)d_in[4];
  const unsigned char* mask    = (const unsigned char*)d_in[5];   // jax bool_ = 1 byte

  float* out  = (float*)d_out;                 // [B,L,D]
  float* attn = out + (size_t)B_ * L_ * D_;    // [B,L,L] (also rel-score scratch)

  rel_scores_kernel<<<dim3(L_ / 64, L_), 128, 0, stream>>>(q, a_key, attn);
  qk_softmax_kernel<<<dim3(L_ / 16, B_), 256, 0, stream>>>(q, k, mask, attn);
  av_kernel       <<<dim3(L_ / 16, B_), 256, 0, stream>>>(attn, v, out);
  rel_av_kernel   <<<L_,               256, 0, stream>>>(attn, a_value, out);
}
